// SinCNN_Origin_82952998355777
// MI455X (gfx1250) — matrix operand order repaired and checked by reading, not compiled, not run on hardware
//
#include <hip/hip_runtime.h>
#include <cstdint>
#include <cstddef>

// ---------------------------------------------------------------------------
// out[b,f] = |  sum_q  e^{i*phi[q]} * c1[(FY[f]-qy)%28] * c1[(FX[f]-qx)%28] * x[b,q] |^2
// with c1 = ifft( exp(-4*pi*i*fftfreq(28)^2) )   (separable WaveSpace kernel)
// => skinny GEMM  X[32768x784] * W^T[784x(10re+10im)]  + |.|^2 epilogue.
// Memory-bound (103 MB / 23.3 TB/s ~ 4.5 us); f16 WMMA w/ f32 accumulate.
// ---------------------------------------------------------------------------

typedef __attribute__((ext_vector_type(16))) _Float16 v16h;
typedef __attribute__((ext_vector_type(8)))  float    v8f;
typedef __attribute__((ext_vector_type(4)))  float    v4f;

#define SZ      28
#define KDIM    784          // 28*28
#define KSTEPS  25           // K padded to 800 = 25*32
#define NFOCUS  10

__device__ __constant__ int c_fy[NFOCUS] = {4, 4, 4, 14, 14, 14, 24, 24, 24, 12};
__device__ __constant__ int c_fx[NFOCUS] = {4, 14, 24, 4, 14, 24, 4, 14, 24, 12};

// ---------------------------------------------------------------------------
// Setup: build Wr/Wi (10x784, zero-padded to 16x800) directly as WMMA
// A fragments, K contiguous per lane half (K = 16h + e), one v16h (32B) per lane per
// K-step.  Total 2 * 25*32*16 halves = 50 KB in d_ws.
// ---------------------------------------------------------------------------
__global__ __launch_bounds__(256) void dnn_setup_kernel(
    const float* __restrict__ phase,
    _Float16* __restrict__ awr,
    _Float16* __restrict__ awi) {
  __shared__ float c1r[SZ], c1i[SZ];
  __shared__ float epr[KDIM], epi[KDIM];

  const int tid = threadIdx.x;
  const float PI = 3.14159265358979323846f;

  // c1[d] = (1/28) * sum_k exp(-4*pi*i*f_k^2) * exp(2*pi*i*k*d/28)
  if (tid < SZ) {
    float sr = 0.f, si = 0.f;
    for (int k = 0; k < SZ; ++k) {
      float fk  = (float)((k < 14) ? k : (SZ - k)) / (float)SZ; // |fftfreq|
      int   m   = (k * tid) % SZ;                               // keep angle small
      float ang = -4.0f * PI * fk * fk + (2.0f * PI / (float)SZ) * (float)m;
      sr += cosf(ang);
      si += sinf(ang);
    }
    c1r[tid] = sr / (float)SZ;
    c1i[tid] = si / (float)SZ;
  }
  for (int q = tid; q < KDIM; q += blockDim.x) {
    float ph = phase[q];
    epr[q] = cosf(ph);
    epi[q] = sinf(ph);
  }
  __syncthreads();

  const int TOT = KSTEPS * 32 * 16;                 // halves per component
  for (int idx = tid; idx < TOT; idx += blockDim.x) {
    int s    = idx >> 9;                            // K-step
    int rem  = idx & 511;
    int lane = rem >> 4;                            // wmma lane
    int e    = rem & 15;                            // v16h element
    int M    = lane & 15;                           // A-matrix row (focus)
    // K order used here (same as B): lanes<16 -> K {0-15}, lanes>=16 -> K {16-31}
    int koff = (lane < 16) ? e : e + 16;
    int q    = s * 32 + koff;

    float vr = 0.f, vi = 0.f;
    if (q < KDIM && M < NFOCUS) {
      int qy = q / SZ, qx = q % SZ;
      int dy = (c_fy[M] - qy + SZ) % SZ;
      int dx = (c_fx[M] - qx + SZ) % SZ;
      // C = c1[dy] * c1[dx]
      float cr = c1r[dy] * c1r[dx] - c1i[dy] * c1i[dx];
      float ci = c1r[dy] * c1i[dx] + c1i[dy] * c1r[dx];
      // W = e^{i*phi[q]} * C
      vr = epr[q] * cr - epi[q] * ci;
      vi = epr[q] * ci + epi[q] * cr;
    }
    awr[idx] = (_Float16)vr;
    awi[idx] = (_Float16)vi;
  }
}

// ---------------------------------------------------------------------------
// Helpers for the GEMM kernel.
// ---------------------------------------------------------------------------
__device__ __forceinline__ v16h cvt16(v4f q0, v4f q1, v4f q2, v4f q3) {
  v16h bf;
  bf[0]  = (_Float16)q0.x; bf[1]  = (_Float16)q0.y;
  bf[2]  = (_Float16)q0.z; bf[3]  = (_Float16)q0.w;
  bf[4]  = (_Float16)q1.x; bf[5]  = (_Float16)q1.y;
  bf[6]  = (_Float16)q1.z; bf[7]  = (_Float16)q1.w;
  bf[8]  = (_Float16)q2.x; bf[9]  = (_Float16)q2.y;
  bf[10] = (_Float16)q2.z; bf[11] = (_Float16)q2.w;
  bf[12] = (_Float16)q3.x; bf[13] = (_Float16)q3.y;
  bf[14] = (_Float16)q3.z; bf[15] = (_Float16)q3.w;
  return bf;
}

// ---------------------------------------------------------------------------
// Main: each wave owns 16 batch rows; 25 K-steps x 2 WMMAs (Re/Im share the
// B fragment).  B fragment: lane n<16 -> row btile+n, K kcol..kcol+15 (64B
// contiguous f32 NT load, cvt to f16); lanes>=16 same rows, next 16 K.
// Tail K-step (s==24, lanes>=16 out of range) is peeled so the main loop is
// branch-free.
// ---------------------------------------------------------------------------
__global__ __launch_bounds__(256) void dnn_gemm_kernel(
    const float* __restrict__ x,
    const _Float16* __restrict__ awr,
    const _Float16* __restrict__ awi,
    float* __restrict__ out,
    int nbatch) {
  const int lane  = threadIdx.x & 31;
  const int wave  = threadIdx.x >> 5;
  const int btile = (blockIdx.x * 8 + wave) * 16;
  if (btile >= nbatch) return;                       // wave-uniform

  const int brow    = btile + (lane & 15);
  const int kcolsel = (lane < 16) ? 0 : 16;
  const float* __restrict__ xrow = x + (size_t)brow * KDIM;

  const v16h* __restrict__ par = (const v16h*)awr;
  const v16h* __restrict__ pai = (const v16h*)awi;

  v8f accr = {};
  v8f acci = {};

  // --- main loop: 24 guard-free K-steps ------------------------------------
  for (int s = 0; s < KSTEPS - 1; ++s) {
    const int kcol = s * 32 + kcolsel;
    const v4f* p = (const v4f*)(xrow + kcol);
    v4f q0 = __builtin_nontemporal_load(p + 0);      // X streamed once: TH=NT
    v4f q1 = __builtin_nontemporal_load(p + 1);
    v4f q2 = __builtin_nontemporal_load(p + 2);
    v4f q3 = __builtin_nontemporal_load(p + 3);
    v16h bf = cvt16(q0, q1, q2, q3);

    v16h ar = par[s * 32 + lane];                    // pre-swizzled A fragments
    v16h ai = pai[s * 32 + lane];

    accr = __builtin_amdgcn_wmma_f32_16x16x32_f16(
        false, ar, false, bf, (short)0, accr, false, false);
    acci = __builtin_amdgcn_wmma_f32_16x16x32_f16(
        false, ai, false, bf, (short)0, acci, false, false);
  }

  // --- peeled tail: s = 24, only lanes<16 (K 768..783) are in range --------
  {
    const int s = KSTEPS - 1;
    v4f q0 = {0.f, 0.f, 0.f, 0.f};
    v4f q1 = q0, q2 = q0, q3 = q0;
    const int kcol = s * 32 + kcolsel;
    if (kcol < KDIM) {
      const v4f* p = (const v4f*)(xrow + kcol);
      q0 = __builtin_nontemporal_load(p + 0);
      q1 = __builtin_nontemporal_load(p + 1);
      q2 = __builtin_nontemporal_load(p + 2);
      q3 = __builtin_nontemporal_load(p + 3);
    }
    v16h bf = cvt16(q0, q1, q2, q3);
    v16h ar = par[s * 32 + lane];                    // zero-padded K>=784
    v16h ai = pai[s * 32 + lane];
    accr = __builtin_amdgcn_wmma_f32_16x16x32_f16(
        false, ar, false, bf, (short)0, accr, false, false);
    acci = __builtin_amdgcn_wmma_f32_16x16x32_f16(
        false, ai, false, bf, (short)0, acci, false, false);
  }

  // D layout: VGPR r, lanes 0-15 -> M=r, N=lane; lanes 16-31 -> M=r+8, N=lane-16
  const int fbase = (lane < 16) ? 0 : 8;
#pragma unroll
  for (int r = 0; r < 8; ++r) {
    int f = fbase + r;
    if (f < NFOCUS) {
      float re = accr[r], im = acci[r];
      out[(size_t)brow * NFOCUS + f] = re * re + im * im;
    }
  }
}

// ---------------------------------------------------------------------------
extern "C" void kernel_launch(void* const* d_in, const int* in_sizes, int n_in,
                              void* d_out, int out_size, void* d_ws, size_t ws_size,
                              hipStream_t stream) {
  const float* x     = (const float*)d_in[0];   // [B,1,28,28] f32
  const float* phase = (const float*)d_in[1];   // [28,28] f32
  float* out = (float*)d_out;                   // [B,10] f32

  _Float16* awr = (_Float16*)d_ws;              // 25*32*16 halves
  _Float16* awi = awr + KSTEPS * 32 * 16;       // 25*32*16 halves (total 50 KB)

  const int nbatch = in_sizes[0] / KDIM;        // 32768

  dnn_setup_kernel<<<1, 256, 0, stream>>>(phase, awr, awi);

  const int nblocks = (nbatch + 127) / 128;     // 8 waves x 16 rows per block
  dnn_gemm_kernel<<<nblocks, 256, 0, stream>>>(x, awr, awi, out, nbatch);
}
